// Attention_16234976379600
// MI455X (gfx1250) — matrix operand order repaired and checked by reading, not compiled, not run on hardware
//
#include <hip/hip_runtime.h>

// ---------------------------------------------------------------------------
// Problem constants (from reference)
// ---------------------------------------------------------------------------
#define S_T    5
#define KH_T   24
#define KW_T   24
#define DIM_C  768
#define HEADS  12
#define HD     64
#define HDE    128           // extended head dim: 64 qk + 24 h + 24 w + 5 t + 11 pad
#define NTOK   2880          // S*KH*KW
#define NF     576           // KH*KW
#define QKV_N  2304          // 3*DIM

typedef __attribute__((ext_vector_type(16))) __bf16 v16bf;
typedef __attribute__((ext_vector_type(8)))  float  v8f;
typedef __attribute__((ext_vector_type(8)))  __bf16 bf16x8;

__device__ __forceinline__ v8f wmma_bf16(v16bf a, v16bf b, v8f c) {
  // D = A(16x32 bf16) * B(32x16 bf16) + C(16x16 f32)
  return __builtin_amdgcn_wmma_f32_16x16x32_bf16(false, a, false, b,
                                                 (short)0, c, false, false);
}

__device__ __forceinline__ v16bf make_frag(bf16x8 lo, bf16x8 hi) {
  return __builtin_shufflevector(lo, hi, 0, 1, 2, 3, 4, 5, 6, 7,
                                 8, 9, 10, 11, 12, 13, 14, 15);
}

__device__ __forceinline__ v8f v8f_zero() {
  v8f z;
#pragma unroll
  for (int e = 0; e < 8; ++e) z[e] = 0.0f;
  return z;
}

// ---------------------------------------------------------------------------
// Elementwise fp32 -> bf16
// ---------------------------------------------------------------------------
__global__ __launch_bounds__(256)
void cvt_bf16_kernel(const float* __restrict__ in, __bf16* __restrict__ out, int n) {
  const int i = blockIdx.x * 256 + threadIdx.x;
  if (i < n) out[i] = (__bf16)in[i];
}

// ---------------------------------------------------------------------------
// Transpose + convert: out[c][r] = (bf16) in[r][c].   R%32==0, C%32==0.
// ---------------------------------------------------------------------------
__global__ __launch_bounds__(256)
void transpose_cvt_kernel(const float* __restrict__ in, __bf16* __restrict__ out,
                          int R, int C) {
  __shared__ __bf16 tile[32][40];
  const int bx = blockIdx.x * 32;
  const int by = blockIdx.y * 32;
  const int tx = threadIdx.x & 31;
  const int ty = threadIdx.x >> 5;
#pragma unroll
  for (int i = 0; i < 32; i += 8)
    tile[ty + i][tx] = (__bf16)in[(size_t)(by + ty + i) * C + bx + tx];
  __syncthreads();
#pragma unroll
  for (int i = 0; i < 32; i += 8)
    out[(size_t)(bx + ty + i) * R + by + tx] = tile[tx][ty + i];
}

// ---------------------------------------------------------------------------
// GEMM: C[M,N] = A[M,K] @ BT[N,K]^T (+ bias[N]).  A,BT bf16; C fp32.
// Block: 128 threads = 4 waves, 64x64 tile; each wave 32x32 (2x2 WMMA).
// ---------------------------------------------------------------------------
template <bool BIAS>
__global__ __launch_bounds__(128)
void gemm_bf16_kernel(const __bf16* __restrict__ A, const __bf16* __restrict__ BT,
                      const float* __restrict__ bias, float* __restrict__ C,
                      int M, int N, int K) {
  __shared__ __bf16 As[64][40];   // [row][k]
  __shared__ __bf16 Bs[64][40];   // [col][k]

  const int t    = threadIdx.x;
  const int lane = t & 31;
  const int wv   = t >> 5;
  const int wm   = (wv >> 1) * 32;
  const int wn   = (wv & 1) * 32;
  const int l16  = lane & 15;
  const int hi   = lane >> 4;

  const int rowBase = blockIdx.y * 64;
  const int colBase = blockIdx.x * 64;

  v8f acc[2][2];
#pragma unroll
  for (int mt = 0; mt < 2; ++mt)
#pragma unroll
    for (int nt = 0; nt < 2; ++nt) acc[mt][nt] = v8f_zero();

  for (int k0 = 0; k0 < K; k0 += 32) {
    {
      const int r = t >> 1, o = (t & 1) * 16;
      const bf16x8* pa = (const bf16x8*)(A + (size_t)(rowBase + r) * K + k0 + o);
      *(bf16x8*)&As[r][o]     = pa[0];
      *(bf16x8*)&As[r][o + 8] = pa[1];
      const bf16x8* pb = (const bf16x8*)(BT + (size_t)(colBase + r) * K + k0 + o);
      *(bf16x8*)&Bs[r][o]     = pb[0];
      *(bf16x8*)&Bs[r][o + 8] = pb[1];
    }
    __syncthreads();

    v16bf afr[2];
#pragma unroll
    for (int mt = 0; mt < 2; ++mt) {
      const int row = wm + mt * 16 + l16;
      afr[mt] = make_frag(*(const bf16x8*)&As[row][hi * 8],
                          *(const bf16x8*)&As[row][16 + hi * 8]);
    }
    v16bf bfr[2];
#pragma unroll
    for (int nt = 0; nt < 2; ++nt) {
      const int col = wn + nt * 16 + l16;
      bfr[nt] = make_frag(*(const bf16x8*)&Bs[col][hi * 8],
                          *(const bf16x8*)&Bs[col][16 + hi * 8]);
    }

#pragma unroll
    for (int mt = 0; mt < 2; ++mt)
#pragma unroll
      for (int nt = 0; nt < 2; ++nt)
        acc[mt][nt] = wmma_bf16(afr[mt], bfr[nt], acc[mt][nt]);

    __syncthreads();
  }

#pragma unroll
  for (int mt = 0; mt < 2; ++mt)
#pragma unroll
    for (int nt = 0; nt < 2; ++nt) {
      const int col = colBase + wn + nt * 16 + l16;
      const float bv = BIAS ? bias[col] : 0.0f;
#pragma unroll
      for (int v = 0; v < 8; ++v) {
        const int row = rowBase + wm + mt * 16 + v + 8 * hi;
        C[(size_t)row * N + col] = acc[mt][nt][v] + bv;
      }
    }
}

// ---------------------------------------------------------------------------
// Split fused-QKV fp32:
//   qe[head][n][0..63]  = q * HD^-0.5 (bf16)   (tail 64..127 filled elsewhere)
//   ke[head][n][0..63]  = k (bf16)             (tail = one-hots, ke_tail kernel)
//   vT[head][hd][n]     = v (bf16, transposed)
// ---------------------------------------------------------------------------
__global__ __launch_bounds__(256)
void pack_qkv_kernel(const float* __restrict__ qkv, __bf16* __restrict__ qe,
                     __bf16* __restrict__ ke, __bf16* __restrict__ vT) {
  const int idx = blockIdx.x * 256 + threadIdx.x;
  if (idx >= NTOK * QKV_N) return;
  const int n = idx / QKV_N;
  const int j = idx - n * QKV_N;
  const int which = j / DIM_C;
  const int head  = (j - which * DIM_C) / HD;
  const int hd    = j & (HD - 1);
  const float val = qkv[idx];
  if (which == 0)
    qe[((size_t)head * NTOK + n) * HDE + hd] = (__bf16)(val * 0.125f);
  else if (which == 1)
    ke[((size_t)head * NTOK + n) * HDE + hd] = (__bf16)val;
  else
    vT[((size_t)head * HD + hd) * NTOK + n] = (__bf16)val;
}

// ---------------------------------------------------------------------------
// Fill ke tail with one-hot key position indicators (bias contraction lanes).
// ---------------------------------------------------------------------------
__global__ __launch_bounds__(256)
void ke_tail_kernel(__bf16* __restrict__ ke) {
  const int idx = blockIdx.x * 256 + threadIdx.x;   // over HEADS*NTOK*64
  if (idx >= HEADS * NTOK * 64) return;
  const int i = idx & 63;
  const int hn = idx >> 6;
  const int n = hn % NTOK;
  const int head = hn / NTOK;
  const int tq = n / NF;
  const int rem = n - tq * NF;
  const int hq = rem / KW_T;
  const int wq = rem - hq * KW_T;
  float v = 0.0f;
  if (i < 24)       v = (i == hq)       ? 1.0f : 0.0f;
  else if (i < 48)  v = (i - 24 == wq)  ? 1.0f : 0.0f;
  else if (i < 53)  v = (i - 48 == tq)  ? 1.0f : 0.0f;
  ke[((size_t)head * NTOK + n) * HDE + 64 + i] = (__bf16)v;
}

// ---------------------------------------------------------------------------
// Rel-pos projections written straight into the qe tail (unscaled q):
//   qe[head][n][64+k]    = sum_c q[c] * rel_pos_h[h(n)-k+23][c]   (k<24)
//   qe[head][n][88+k]    = rw ..., qe[head][n][112+k] = rt ..., pad zeros.
// ---------------------------------------------------------------------------
__global__ __launch_bounds__(64)
void rel_tables_kernel(const float* __restrict__ qkv,
                       const float* __restrict__ rph, const float* __restrict__ rpw,
                       const float* __restrict__ rpt, __bf16* __restrict__ qe) {
  __shared__ float qrow[HD];
  const int n = blockIdx.x, head = blockIdx.y, t = threadIdx.x;
  qrow[t] = qkv[(size_t)n * QKV_N + head * HD + t];
  __syncthreads();

  const int tq = n / NF;
  const int rem = n - tq * NF;
  const int hq = rem / KW_T;
  const int wq = rem - hq * KW_T;

  float s = 0.0f;
  if (t < 24) {
    const float* r = rph + (size_t)(hq - t + (KH_T - 1)) * HD;
#pragma unroll 8
    for (int c = 0; c < HD; ++c) s += qrow[c] * r[c];
  } else if (t < 48) {
    const float* r = rpw + (size_t)(wq - (t - 24) + (KW_T - 1)) * HD;
#pragma unroll 8
    for (int c = 0; c < HD; ++c) s += qrow[c] * r[c];
  } else if (t < 53) {
    const float* r = rpt + (size_t)(tq - (t - 48) + (S_T - 1)) * HD;
#pragma unroll 8
    for (int c = 0; c < HD; ++c) s += qrow[c] * r[c];
  }
  qe[((size_t)head * NTOK + n) * HDE + 64 + t] = (__bf16)s;  // t>=53 -> 0
}

// ---------------------------------------------------------------------------
// Flash attention with bias folded into the QK^T contraction (HDE=128).
// Block = (head, 64-query tile), 4 waves x 16 queries.
// All WMMA fragment reads are contiguous ds_load_b128 pairs.
// ---------------------------------------------------------------------------
__global__ __launch_bounds__(128)
void flash_attn_kernel(const __bf16* __restrict__ qe, const __bf16* __restrict__ ke,
                       const __bf16* __restrict__ vT, __bf16* __restrict__ out) {
  __shared__ __bf16 Qs[64][136];   // [query][128 ext-dim]
  __shared__ __bf16 Ks[64][136];   // [key][128 ext-dim]
  __shared__ __bf16 VsT[64][72];   // [hd][key]
  __shared__ __bf16 Ps[4][16][72]; // per-wave P strip [query][key]

  const int t    = threadIdx.x;
  const int lane = t & 31;
  const int wv   = t >> 5;
  const int l16  = lane & 15;
  const int hi   = lane >> 4;
  const int head = blockIdx.y;
  const int n0   = blockIdx.x * 64;
  const size_t ebase = (size_t)head * NTOK * HDE;

  // ---- load Q tile (scale + bias lanes prebaked) ----
  {
    const int r = t >> 1, o = (t & 1) * 64;
    const bf16x8* src = (const bf16x8*)(qe + ebase + (size_t)(n0 + r) * HDE + o);
#pragma unroll
    for (int i = 0; i < 8; ++i) *(bf16x8*)&Qs[r][o + i * 8] = src[i];
  }
  __syncthreads();

  // ---- preload Q A-fragments (16 queries x 128 dims, 4 K-chunks) ----
  v16bf aq[4];
#pragma unroll
  for (int kc = 0; kc < 4; ++kc) {
    const int row = wv * 16 + l16;
    aq[kc] = make_frag(*(const bf16x8*)&Qs[row][kc * 32 + hi * 8],
                       *(const bf16x8*)&Qs[row][kc * 32 + 16 + hi * 8]);
  }

  v8f oac[4];
#pragma unroll
  for (int nt = 0; nt < 4; ++nt) oac[nt] = v8f_zero();
  float mrow[8], lrow[8];
#pragma unroll
  for (int v = 0; v < 8; ++v) { mrow[v] = -1e30f; lrow[v] = 0.0f; }

  for (int kt = 0; kt < NTOK / 64; ++kt) {
    const int kk0 = kt * 64;
    // ---- stage K-ext [key][128] and V^T [hd][key] tiles ----
    {
      const int r = t >> 1, o = (t & 1) * 64;
      const bf16x8* ks = (const bf16x8*)(ke + ebase + (size_t)(kk0 + r) * HDE + o);
#pragma unroll
      for (int i = 0; i < 8; ++i) *(bf16x8*)&Ks[r][o + i * 8] = ks[i];
      const int o2 = (t & 1) * 32;
      const bf16x8* vs =
          (const bf16x8*)(vT + ((size_t)head * HD + r) * NTOK + kk0 + o2);
#pragma unroll
      for (int i = 0; i < 4; ++i) *(bf16x8*)&VsT[r][o2 + i * 8] = vs[i];
    }
    __syncthreads();

    // ---- S = Qe @ Ke^T  (16 queries x 64 keys, logits + bias in one pass) ----
    v8f sac[4];
#pragma unroll
    for (int nt = 0; nt < 4; ++nt) {
      sac[nt] = v8f_zero();
#pragma unroll
      for (int kc = 0; kc < 4; ++kc) {
        const int keyr = nt * 16 + l16;
        const v16bf bk =
            make_frag(*(const bf16x8*)&Ks[keyr][kc * 32 + hi * 8],
                      *(const bf16x8*)&Ks[keyr][kc * 32 + 16 + hi * 8]);
        sac[nt] = wmma_bf16(aq[kc], bk, sac[nt]);
      }
    }

    // ---- online softmax (rows live across 16 lanes) ----
    float sreg[4][8];
#pragma unroll
    for (int v = 0; v < 8; ++v) {
      float mx = fmaxf(fmaxf(sac[0][v], sac[1][v]), fmaxf(sac[2][v], sac[3][v]));
#pragma unroll
      for (int off = 1; off < 16; off <<= 1) mx = fmaxf(mx, __shfl_xor(mx, off, 16));
      const float mnew  = fmaxf(mrow[v], mx);
      const float alpha = __expf(mrow[v] - mnew);
      float sum = 0.0f;
#pragma unroll
      for (int nt = 0; nt < 4; ++nt) {
        const float p = __expf(sac[nt][v] - mnew);
        sreg[nt][v] = p;
        sum += p;
      }
#pragma unroll
      for (int off = 1; off < 16; off <<= 1) sum += __shfl_xor(sum, off, 16);
      lrow[v] = lrow[v] * alpha + sum;
      mrow[v] = mnew;
#pragma unroll
      for (int nt = 0; nt < 4; ++nt) {
        oac[nt][v] *= alpha;
        Ps[wv][v + 8 * hi][nt * 16 + l16] = (__bf16)sreg[nt][v];  // C->A relayout
      }
    }

    // ---- O += P @ V ----
#pragma unroll
    for (int kc = 0; kc < 2; ++kc) {
      const v16bf ap =
          make_frag(*(const bf16x8*)&Ps[wv][l16][kc * 32 + hi * 8],
                    *(const bf16x8*)&Ps[wv][l16][kc * 32 + 16 + hi * 8]);
#pragma unroll
      for (int nt = 0; nt < 4; ++nt) {
        const int hdr = nt * 16 + l16;    // B[key][hd] = V^T[hd][key]
        const v16bf bv =
            make_frag(*(const bf16x8*)&VsT[hdr][kc * 32 + hi * 8],
                      *(const bf16x8*)&VsT[hdr][kc * 32 + 16 + hi * 8]);
        oac[nt] = wmma_bf16(ap, bv, oac[nt]);
      }
    }
    __syncthreads();  // before next tile overwrites Ks/VsT
  }

  // ---- normalize and store bf16 [n][head*64+hd] ----
#pragma unroll
  for (int v = 0; v < 8; ++v) {
    const float inv = 1.0f / lrow[v];
    const int q = n0 + wv * 16 + v + 8 * hi;
#pragma unroll
    for (int nt = 0; nt < 4; ++nt) {
      const int c = head * HD + nt * 16 + l16;
      out[(size_t)q * DIM_C + c] = (__bf16)(oac[nt][v] * inv);
    }
  }
}

// ---------------------------------------------------------------------------
// Launch
// ---------------------------------------------------------------------------
extern "C" void kernel_launch(void* const* d_in, const int* in_sizes, int n_in,
                              void* d_out, int out_size, void* d_ws, size_t ws_size,
                              hipStream_t stream) {
  const float* x      = (const float*)d_in[0];
  const float* w_qkv  = (const float*)d_in[1];
  const float* w_proj = (const float*)d_in[2];
  const float* b_proj = (const float*)d_in[3];
  const float* rph    = (const float*)d_in[4];
  const float* rpw    = (const float*)d_in[5];
  const float* rpt    = (const float*)d_in[6];
  float* out = (float*)d_out;

  char* ws = (char*)d_ws;
  size_t off = 0;
  float*  qkvf = (float*)(ws + off);  off += (size_t)NTOK * QKV_N * 4;        // 26.5 MB
  __bf16* xb   = (__bf16*)(ws + off); off += (size_t)NTOK * DIM_C * 2;        // 4.4 MB
  __bf16* wqT  = (__bf16*)(ws + off); off += (size_t)QKV_N * DIM_C * 2;       // 3.5 MB
  __bf16* wpT  = (__bf16*)(ws + off); off += (size_t)DIM_C * DIM_C * 2;       // 1.2 MB
  __bf16* qeb  = (__bf16*)(ws + off); off += (size_t)HEADS * NTOK * HDE * 2;  // 8.8 MB
  __bf16* keb  = (__bf16*)(ws + off); off += (size_t)HEADS * NTOK * HDE * 2;  // 8.8 MB
  __bf16* vT   = (__bf16*)(ws + off); off += (size_t)HEADS * NTOK * HD * 2;   // 4.4 MB
  __bf16* aout = (__bf16*)(ws + off); off += (size_t)NTOK * DIM_C * 2;        // 4.4 MB

  // 0) one-time precision/layout prep
  cvt_bf16_kernel<<<(NTOK * DIM_C + 255) / 256, 256, 0, stream>>>(x, xb,
                                                                  NTOK * DIM_C);
  transpose_cvt_kernel<<<dim3(QKV_N / 32, DIM_C / 32), 256, 0, stream>>>(
      w_qkv, wqT, DIM_C, QKV_N);
  transpose_cvt_kernel<<<dim3(DIM_C / 32, DIM_C / 32), 256, 0, stream>>>(
      w_proj, wpT, DIM_C, DIM_C);

  // 1) QKV projection
  gemm_bf16_kernel<false><<<dim3(QKV_N / 64, NTOK / 64), 128, 0, stream>>>(
      xb, wqT, nullptr, qkvf, NTOK, QKV_N, DIM_C);
  // 2) pack Q/K into extended-dim buffers, V transposed
  pack_qkv_kernel<<<(NTOK * QKV_N + 255) / 256, 256, 0, stream>>>(qkvf, qeb, keb, vT);
  // 2b) one-hot bias lanes of Ke
  ke_tail_kernel<<<(HEADS * NTOK * 64 + 255) / 256, 256, 0, stream>>>(keb);
  // 3) rel-pos projections into Qe tail
  rel_tables_kernel<<<dim3(NTOK, HEADS), 64, 0, stream>>>(qkvf, rph, rpw, rpt, qeb);
  // 4) fused flash attention (bias inside the WMMA contraction)
  flash_attn_kernel<<<dim3(NTOK / 64, HEADS), 128, 0, stream>>>(qeb, keb, vT, aout);
  // 5) output projection (+bias)
  gemm_bf16_kernel<true><<<dim3(DIM_C / 64, NTOK / 64), 128, 0, stream>>>(
      aout, wpT, b_proj, out, NTOK, DIM_C, DIM_C);
}
